// SequenceDecoderLSTM_37400575213990
// MI455X (gfx1250) — matrix operand order repaired and checked by reading, not compiled, not run on hardware
//
#include <hip/hip_runtime.h>
#include <hip/hip_bf16.h>
#include <math.h>

// ---------------- problem constants ----------------
#define HDIM 256          // hidden size
#define GDIM 1024         // 4*h
#define BTOT 4096         // total batch rows (64*64)
#define TSTEPS 50         // sequence length
#define ROWS_PER_WG 128   // 8 waves * 16 rows
#define CHUNK_ROWS 80     // 64 W_hh rows + 16 W_out rows per column-tile
#define ROW_BYTES 512     // payload bytes per weight row (256 bf16)
#define ROW_PITCH 528     // padded row pitch in LDS (bank-conflict free, 16B aligned)
#define HROW_ELEMS 264    // hbuf row pitch in bf16 elements (528 B)
#define CHUNK_BYTES (CHUNK_ROWS * ROW_PITCH)   // 42240 B per padded chunk
#define UNITS_PER_THREAD 10                    // 80 rows * 32 units / 256 threads

typedef __attribute__((ext_vector_type(16))) __bf16 v16bf;
typedef __attribute__((ext_vector_type(8)))  float  v8f;
typedef __attribute__((ext_vector_type(4)))  int    v4i;

// ---------------- CDNA5 feature detection ----------------
#if defined(__has_builtin)
#  if __has_builtin(__builtin_amdgcn_global_load_async_to_lds_b128)
#    define HAVE_ASYNC_LDS 1
#  endif
#  if __has_builtin(__builtin_amdgcn_s_wait_asynccnt)
#    define HAVE_WAIT_ASYNC 1
#  endif
#  if __has_builtin(__builtin_amdgcn_tanhf)
#    define TANHF(x) __builtin_amdgcn_tanhf(x)
#  endif
#endif
#ifndef TANHF
#  define TANHF(x) tanhf(x)
#endif

#define WMMA_BF16(a, b, c) \
  __builtin_amdgcn_wmma_f32_16x16x32_bf16(false, (a), false, (b), (short)0, (c), false, false)

__device__ __forceinline__ float sigf(float x) { return 1.0f / (1.0f + __expf(-x)); }

__device__ __forceinline__ void wait_async_10() {
#ifdef HAVE_ASYNC_LDS
#  ifdef HAVE_WAIT_ASYNC
  __builtin_amdgcn_s_wait_asynccnt(10);
#  else
  asm volatile("s_wait_asynccnt 10" ::: "memory");
#  endif
#endif
}

// ---------------- kernel 1: weight conversion to bf16 ----------------
__global__ __launch_bounds__(256) void conv_weights(const float* __restrict__ whh,
                                                    const float* __restrict__ wout,
                                                    __bf16* __restrict__ whh_bf,
                                                    __bf16* __restrict__ wout_bf) {
  int i = blockIdx.x * 256 + threadIdx.x;
  if (i < GDIM * HDIM) whh_bf[i] = (__bf16)whh[i];
  if (i < HDIM * HDIM) wout_bf[i] = (__bf16)wout[i];
}

// ---------------- kernel 2: x_proj = h0 @ W_ih^T + b_ih + b_hh, packed ----------------
// Packed layout: xp[((mtile*16 + j0)*4 + g)*256 + lane*8 + reg]  (C-fragment order)
__global__ __launch_bounds__(256) void xproj_pack(const float* __restrict__ x,
                                                  const float* __restrict__ Wih,
                                                  const float* __restrict__ bih,
                                                  const float* __restrict__ bhh,
                                                  float* __restrict__ xp) {
  __shared__ float xs[16][HDIM];
  const int t  = threadIdx.x;
  const int mt = blockIdx.x;  // 0..255 (tile of 16 batch rows)
  for (int i = 0; i < 16; ++i) {
    int idx = t + 256 * i;
    int row = idx >> 8, col = idx & 255;
    xs[row][col] = x[(mt * 16 + row) * HDIM + col];
  }
  __syncthreads();

  float acc[4][16];
#pragma unroll
  for (int g = 0; g < 4; ++g)
#pragma unroll
    for (int r = 0; r < 16; ++r) acc[g][r] = 0.0f;

  for (int k = 0; k < HDIM; ++k) {
    float w[4];
#pragma unroll
    for (int g = 0; g < 4; ++g) w[g] = Wih[(g * HDIM + t) * HDIM + k];
#pragma unroll
    for (int r = 0; r < 16; ++r) {
      float xv = xs[r][k];
#pragma unroll
      for (int g = 0; g < 4; ++g) acc[g][r] += w[g] * xv;
    }
  }

  const int j0 = t >> 4, nc = t & 15;
#pragma unroll
  for (int g = 0; g < 4; ++g) {
    float bias = bih[g * HDIM + t] + bhh[g * HDIM + t];
#pragma unroll
    for (int r = 0; r < 16; ++r) {
      int lane = nc + 16 * (r >> 3);
      int reg  = r & 7;
      xp[(((mt * 16 + j0) * 4 + g) * 256) + lane * 8 + reg] = acc[g][r] + bias;
    }
  }
}

// ---------------- kernel 3: fused 50-step LSTM + output projection ----------------
__global__ __launch_bounds__(256, 1) void lstm_fused(const float* __restrict__ x,
                                                     const float* __restrict__ xp,
                                                     const __bf16* __restrict__ whh_bf,
                                                     const __bf16* __restrict__ wout_bf,
                                                     const float* __restrict__ c0,
                                                     const float* __restrict__ bout,
                                                     float* __restrict__ out) {
  // 3 ring chunks (padded rows) + padded hidden + f32 cell + bias cache = ~319KB LDS
  __shared__ __bf16 chunkbuf[3 * CHUNK_ROWS * HROW_ELEMS];   // 126720 B
  __shared__ __bf16 hbuf[ROWS_PER_WG][HROW_ELEMS];           // 67584 B
  __shared__ float  cbuf[8][16][32][8];                      // 131072 B, wave-private
  __shared__ float  bout_l[HDIM];                            // 1024 B

  const int tid   = threadIdx.x;
  const int wave  = tid >> 5;
  const int lane  = tid & 31;
  const int half  = lane >> 4;
  const int ln    = lane & 15;
  const int lrow0 = wave * 16;
  const int m0    = blockIdx.x * ROWS_PER_WG + lrow0;  // first global batch row of this wave
  const int mtile = m0 >> 4;

  // ---- init h0 = x (bf16) into LDS; cache b_out ----
  for (int r = 0; r < 16; ++r)
    for (int cc = 0; cc < 8; ++cc) {
      int col = lane + 32 * cc;
      hbuf[lrow0 + r][col] = (__bf16)x[(m0 + r) * HDIM + col];
    }
  bout_l[tid] = bout[tid];

  // ---- init cell state in LDS (c0 broadcast across rows) ----
#pragma unroll
  for (int j = 0; j < 16; ++j) {
    float cv = c0[j * 16 + ln];
    float4 q = {cv, cv, cv, cv};
    *(float4*)&cbuf[wave][j][lane][0] = q;
    *(float4*)&cbuf[wave][j][lane][4] = q;
  }

  // A-fragment loader: h rows of this wave, documented 16-bit A layout
  // lane<16: K={32k..32k+7, 32k+16..23}; lane>=16: K={32k+8..15, 32k+24..31}
  auto loadA = [&](int k) -> v16bf {
    union { v16bf v; int4 q[2]; } u;
    const char* rowp = (const char*)&hbuf[0][0] + (lrow0 + ln) * ROW_PITCH;
    int base = (32 * k + 8 * half) * 2;
    u.q[0] = *(const int4*)(rowp + base);
    u.q[1] = *(const int4*)(rowp + base + 32);
    return u.v;
  };
  // B-fragment loader from a staged chunk: lane ln holds weight row (rowbase+ln),
  // lane<16 -> K={32k..32k+7, 32k+16..23}; lane>=16 -> K={32k+8..15, 32k+24..31}
  auto loadB = [&](const char* cbase, int rowbase, int k) -> v16bf {
    union { v16bf v; int4 q[2]; } u;
    const char* rowp = cbase + (rowbase + ln) * ROW_PITCH;
    int base = (32 * k + 8 * half) * 2;
    u.q[0] = *(const int4*)(rowp + base);
    u.q[1] = *(const int4*)(rowp + base + 32);
    return u.v;
  };
  // Stage one chunk for column-tile jn into ring slot `buf` (padded row pitch).
  // Rows 0..63 = W_hh (4 gates x 16 rows), rows 64..79 = W_out rows.
  auto stage = [&](int buf, int jn) {
    char* dstbase = (char*)chunkbuf + (size_t)buf * CHUNK_BYTES;
#pragma unroll
    for (int i = 0; i < UNITS_PER_THREAD; ++i) {
      int u     = tid + 256 * i;
      int cr    = u >> 5;          // chunk row (32 x 16B units per row), 0..79
      int inrow = (u & 31) * 16;   // byte offset inside 512B payload
      const char* src;
      if (cr < 64) {
        int g = cr >> 4, r = cr & 15;
        src = (const char*)whh_bf + (size_t)(g * HDIM + jn * 16 + r) * ROW_BYTES + inrow;
      } else {
        int r = cr - 64;
        src = (const char*)wout_bf + (size_t)(jn * 16 + r) * ROW_BYTES + inrow;
      }
      char* dst = dstbase + cr * ROW_PITCH + inrow;
#ifdef HAVE_ASYNC_LDS
      __builtin_amdgcn_global_load_async_to_lds_b128(
          (__attribute__((address_space(1))) v4i*)(src),
          (__attribute__((address_space(3))) v4i*)(dst), 0, 0);
#else
      *(int4*)(dst) = *(const int4*)(src);
#endif
    }
  };

  v16bf afr[8];
#pragma unroll
  for (int k = 0; k < 8; ++k) afr[k] = loadA(k);

  stage(0, 0);      // prologue: chunk for column-tile 0 into ring slot 0
  int cur = 0;      // ring slot holding the chunk we consume this phase

#pragma unroll 1
  for (int t = 0; t <= TSTEPS; ++t) {
#pragma unroll 1
    for (int j0 = 0; j0 < 16; ++j0) {
      int nxt = (cur == 2) ? 0 : cur + 1;
      // Ring-3 => a wave issuing stage(p+1) has passed barrier(p-1), which every
      // wave only signals after finishing compute(p-2) = last reader of slot nxt.
      stage(nxt, (j0 + 1) & 15);
      wait_async_10();                 // my loads for chunk j0 are complete
      __syncthreads();                 // chunk j0 visible WG-wide (single barrier/phase)
      const char* cb = (const char*)chunkbuf + (size_t)cur * CHUNK_BYTES;

      if (t > 0) {
        // output projection of h_{t-1} (afr still holds it), cols j0*16..+15
        v8f oacc;
        float bv = bout_l[j0 * 16 + ln];
#pragma unroll
        for (int r = 0; r < 8; ++r) oacc[r] = bv;
        v16bf bw = loadB(cb, 64, 0);
#pragma unroll
        for (int k = 0; k < 8; ++k) {
          v16bf bn;
          if (k < 7) bn = loadB(cb, 64, k + 1);
          oacc = WMMA_BF16(afr[k], bw, oacc);
          if (k < 7) bw = bn;
        }
#pragma unroll
        for (int r = 0; r < 8; ++r) {
          int m = m0 + r + 8 * half;
          out[((size_t)m * TSTEPS + (t - 1)) * HDIM + j0 * 16 + ln] = oacc[r];
        }
      }

      if (t < TSTEPS) {
        // gate GEMM tiles (i,f,g,o) for hidden columns j0*16..+15
        union { v8f v; float4 q[2]; } u0, u1, u2, u3;
        const float* xpj = xp + (size_t)((mtile * 16 + j0) * 4) * 256;
        u0.q[0] = *(const float4*)(xpj + 0 * 256 + lane * 8);
        u0.q[1] = *(const float4*)(xpj + 0 * 256 + lane * 8 + 4);
        u1.q[0] = *(const float4*)(xpj + 1 * 256 + lane * 8);
        u1.q[1] = *(const float4*)(xpj + 1 * 256 + lane * 8 + 4);
        u2.q[0] = *(const float4*)(xpj + 2 * 256 + lane * 8);
        u2.q[1] = *(const float4*)(xpj + 2 * 256 + lane * 8 + 4);
        u3.q[0] = *(const float4*)(xpj + 3 * 256 + lane * 8);
        u3.q[1] = *(const float4*)(xpj + 3 * 256 + lane * 8 + 4);
        v8f ai = u0.v, af_ = u1.v, ag = u2.v, ao = u3.v;

        // software-pipelined: issue k+1 fragments before consuming k's
        v16bf bi = loadB(cb, 0, 0), bf = loadB(cb, 16, 0),
              bg = loadB(cb, 32, 0), bo = loadB(cb, 48, 0);
#pragma unroll
        for (int k = 0; k < 8; ++k) {
          v16bf bi_n, bf_n, bg_n, bo_n;
          if (k < 7) {
            bi_n = loadB(cb, 0,  k + 1);
            bf_n = loadB(cb, 16, k + 1);
            bg_n = loadB(cb, 32, k + 1);
            bo_n = loadB(cb, 48, k + 1);
          }
          ai  = WMMA_BF16(afr[k], bi, ai);
          af_ = WMMA_BF16(afr[k], bf, af_);
          ag  = WMMA_BF16(afr[k], bg, ag);
          ao  = WMMA_BF16(afr[k], bo, ao);
          if (k < 7) { bi = bi_n; bf = bf_n; bg = bg_n; bo = bo_n; }
        }

        // fused activations + cell/hidden update (cell state in LDS)
        float* cp = &cbuf[wave][j0][lane][0];
        union { float a[8]; float4 q[2]; } cu;
        cu.q[0] = *(const float4*)(cp);
        cu.q[1] = *(const float4*)(cp + 4);
#pragma unroll
        for (int r = 0; r < 8; ++r) {
          float iv = sigf(ai[r]);
          float fv = sigf(af_[r]);
          float gv = TANHF(ag[r]);
          float ov = sigf(ao[r]);
          float cn = fv * cu.a[r] + iv * gv;
          cu.a[r] = cn;
          float hn = ov * TANHF(cn);
          hbuf[lrow0 + r + 8 * half][j0 * 16 + ln] = (__bf16)hn;
        }
        *(float4*)(cp)     = cu.q[0];
        *(float4*)(cp + 4) = cu.q[1];
      }
      cur = nxt;
    }
    if (t < TSTEPS) {
#pragma unroll
      for (int k = 0; k < 8; ++k) afr[k] = loadA(k);  // reload A-frags of h_t
    }
  }
}

// ---------------- launch ----------------
extern "C" void kernel_launch(void* const* d_in, const int* in_sizes, int n_in,
                              void* d_out, int out_size, void* d_ws, size_t ws_size,
                              hipStream_t stream) {
  const float* x    = (const float*)d_in[0];
  const float* Wih  = (const float*)d_in[1];
  const float* Whh  = (const float*)d_in[2];
  const float* bih  = (const float*)d_in[3];
  const float* bhh  = (const float*)d_in[4];
  const float* c0   = (const float*)d_in[5];
  const float* Wout = (const float*)d_in[6];
  const float* bout = (const float*)d_in[7];
  float* out = (float*)d_out;

  char* wsb = (char*)d_ws;
  float*  xp      = (float*)wsb;                              // 16 MB packed x_proj
  __bf16* whh_bf  = (__bf16*)(wsb + 16777216);                // 512 KB
  __bf16* wout_bf = (__bf16*)(wsb + 16777216 + 524288);       // 128 KB

  conv_weights<<<1024, 256, 0, stream>>>(Whh, Wout, whh_bf, wout_bf);
  xproj_pack<<<256, 256, 0, stream>>>(x, Wih, bih, bhh, xp);
  lstm_fused<<<32, 256, 0, stream>>>(x, xp, whh_bf, wout_bf, c0, bout, out);
}
